// BioJepa_1778116460853
// MI455X (gfx1250) — matrix operand order repaired and checked by reading, not compiled, not run on hardware
//
#include <hip/hip_runtime.h>
#include <hip/hip_bf16.h>
#include <math.h>

typedef __attribute__((ext_vector_type(16))) _Float16 v16h;
typedef __attribute__((ext_vector_type(8)))  _Float16 v8h;
typedef __attribute__((ext_vector_type(4)))  _Float16 v4h;
typedef __attribute__((ext_vector_type(2)))  _Float16 v2h;
typedef __attribute__((ext_vector_type(2)))  __fp16   v2fp16;
typedef __attribute__((ext_vector_type(8)))  float    v8f;

namespace {
constexpr int Bc  = 8;
constexpr int CLc = 2048;
constexpr int TTc = 1024;
constexpr int Dc  = 768;
constexpr int Acn = 320;
constexpr int Lc  = 6;
constexpr int Hc  = 12;
constexpr int Fc  = 4 * Dc;        // 3072
constexpr int Tc  = CLc + TTc;     // 3072
constexpr int NTOK = Bc * Tc;      // 24576
constexpr int HD  = Dc / Hc;       // 64
}

__device__ __forceinline__ float elu1f(float x) {
  return x > 0.f ? x + 1.f : expf(x);        // elu(x)+1
}
__device__ __forceinline__ float gelu_tanhf(float x) {
  float c = 0.7978845608028654f * (x + 0.044715f * x * x * x);
  return 0.5f * x * (1.f + tanhf(c));
}
__device__ __forceinline__ float gelu_exactf(float x) {
  return 0.5f * x * (1.f + erff(x * 0.70710678118654752f));
}

// packed f32->f16 (v_cvt_pk_rtz_f16_f32); builtin returns __fp16 vec -> bit-cast
__device__ __forceinline__ v2h pk2(float a, float b) {
  const v2fp16 r = __builtin_amdgcn_cvt_pkrtz(a, b);
  return __builtin_bit_cast(v2h, r);
}
__device__ __forceinline__ v4h pk4(float a, float b, float c, float d) {
  const v2h lo = pk2(a, b);
  const v2h hi = pk2(c, d);
  return __builtin_shufflevector(lo, hi, 0, 1, 2, 3);
}
// fragment = two contiguous 16B LDS chunks (2x ds_load_b128)
__device__ __forceinline__ v16h make_frag(const _Float16* lo, const _Float16* hi) {
  const v8h a = *(const v8h*)lo;
  const v8h b = *(const v8h*)hi;
  return __builtin_shufflevector(a, b, 0,1,2,3,4,5,6,7,8,9,10,11,12,13,14,15);
}

// CDNA5 async global->LDS copy (ASYNCcnt path). LDS byte offset = low 32 bits
// of the generic shared-memory pointer (flat aperture rule).
__device__ __forceinline__ void async_copy_b128(void* lds_ptr, const void* gptr) {
  const unsigned lds_off = (unsigned)(unsigned long long)lds_ptr;
  asm volatile("global_load_async_to_lds_b128 %0, %1, off"
               :: "v"(lds_off), "v"(gptr) : "memory");
}
__device__ __forceinline__ void wait_async0() {
  asm volatile("s_wait_asynccnt 0" ::: "memory");
}

// ---------------------------------------------------------------------------
// Generic WMMA GEMM: C[M,N] = epi(A[M,K](f16) @ W[K,N](f32) + bias[N])
// EPI: 0 bias | 1 bias+elu+1 | 2 bias+gelu(tanh) | 3 bias+residual | 4 bias+clip
// 128x128x32 tile, 256 thr = 8 waves (2Mx4N), 4x2 WMMA tiles per wave.
// A (f16, k-contiguous) staged with GLOBAL_LOAD_ASYNC_TO_LDS_B128; W tile
// converted f32->f16 and stored TRANSPOSED (n-major) so both fragments are
// 2x ds_load_b128 per the ISA f16 operand layouts.
// ---------------------------------------------------------------------------
template<int EPI, typename OutT>
__global__ __launch_bounds__(256) void gemm_kernel(const _Float16* __restrict__ A,
                                                   const float* __restrict__ W,
                                                   const float* __restrict__ bias,
                                                   OutT* __restrict__ C,
                                                   int M, int N, int K)
{
  __shared__ _Float16 As[128][40];    // [m][k]
  __shared__ _Float16 BsT[128][40];   // [n][k]

  const int tid   = threadIdx.x;
  const int lane  = tid & 31;
  const int w     = tid >> 5;
  const int wm    = w >> 2;           // 0..1
  const int wn    = w & 3;            // 0..3
  const int bm    = blockIdx.y * 128;
  const int bn    = blockIdx.x * 128;
  const int mlo   = lane & 15;
  const int khalf = lane >> 4;

  const v8f vzero = {0.f,0.f,0.f,0.f,0.f,0.f,0.f,0.f};
  v8f acc[4][2];
  #pragma unroll
  for (int i = 0; i < 4; ++i) { acc[i][0] = vzero; acc[i][1] = vzero; }

  for (int k0 = 0; k0 < K; k0 += 32) {
    if (k0 + 32 < K) {   // next W tile toward L2 (global_prefetch_b8)
      __builtin_prefetch(W + (size_t)(k0 + 32 + (tid >> 3)) * N + bn, 0, 1);
    }
    // A tile: 128x32 f16 = 8KB, raw async copy, 2x b128 per thread
    #pragma unroll
    for (int j = 0; j < 2; ++j) {
      const int s  = tid + j * 256;            // 512 slots of 8 halves
      const int ra = s >> 2, ca = (s & 3) << 3;
      async_copy_b128(&As[ra][ca], A + (size_t)(bm + ra) * K + (k0 + ca));
    }
    // W tile: gather k-quads of column (bn+nb), convert, store transposed
    #pragma unroll
    for (int j = 0; j < 4; ++j) {
      const int s  = tid + j * 256;            // 1024 quad-slots
      const int nb = s & 127, kq = (s >> 7) << 2;
      const float* wp = W + (size_t)(k0 + kq) * N + (bn + nb);
      *(v4h*)&BsT[nb][kq] = pk4(wp[0], wp[(size_t)N], wp[2 * (size_t)N], wp[3 * (size_t)N]);
    }
    wait_async0();
    __syncthreads();

    v16h af[4], bf[2];
    const int kb  = khalf * 8;
    const int kb2 = khalf * 16;
    #pragma unroll
    for (int mt = 0; mt < 4; ++mt) {
      const int m = wm * 64 + mt * 16 + mlo;
      af[mt] = make_frag(&As[m][kb], &As[m][kb + 16]);
    }
    #pragma unroll
    for (int nt = 0; nt < 2; ++nt) {
      const int n = wn * 32 + nt * 16 + mlo;
      bf[nt] = make_frag(&BsT[n][kb], &BsT[n][kb + 16]);
    }
    #pragma unroll
    for (int mt = 0; mt < 4; ++mt)
      #pragma unroll
      for (int nt = 0; nt < 2; ++nt)
        acc[mt][nt] = __builtin_amdgcn_wmma_f32_16x16x32_f16(
            false, af[mt], false, bf[nt], (short)0, acc[mt][nt], false, false);
    __syncthreads();
  }

  #pragma unroll
  for (int mt = 0; mt < 4; ++mt)
    #pragma unroll
    for (int nt = 0; nt < 2; ++nt)
      #pragma unroll
      for (int r = 0; r < 8; ++r) {
        const int row = bm + wm * 64 + mt * 16 + khalf * 8 + r;
        const int col = bn + wn * 32 + nt * 16 + mlo;
        float v = acc[mt][nt][r] + bias[col];
        if      (EPI == 1) v = elu1f(v);
        else if (EPI == 2) v = gelu_tanhf(v);
        else if (EPI == 3) v = v + (float)C[(size_t)row * N + col];
        else if (EPI == 4) v = fminf(fmaxf(v, -10.f), 2.f);
        C[(size_t)row * N + col] = (OutT)v;
      }
}

// ---------------------------------------------------------------------------
// Self-attention kv moment: kvm[b,h] = sum_t k[t]^T v[t]  (64x64, K=Tc)
// grid.x = B*H, 128 threads (4 waves). k/v tiles stored [d][t] in LDS so both
// fragments are k(t)-contiguous b128 loads.
// ---------------------------------------------------------------------------
__global__ __launch_bounds__(128) void attn_kv_kernel(const float* __restrict__ kbuf,
                                                      const float* __restrict__ vbuf,
                                                      float* __restrict__ kvm)
{
  __shared__ _Float16 ksT[64][40];   // [d][t_local]
  __shared__ _Float16 vsT[64][40];   // [d][t_local]
  const int bh = blockIdx.x;
  const int b  = bh / Hc, h = bh % Hc;
  const size_t base = ((size_t)b * Tc) * Dc + h * HD;
  const int tid = threadIdx.x, lane = tid & 31, w = tid >> 5;
  const int mlo = lane & 15, khalf = lane >> 4;

  const v8f vzero = {0.f,0.f,0.f,0.f,0.f,0.f,0.f,0.f};
  v8f acc[4];
  #pragma unroll
  for (int i = 0; i < 4; ++i) acc[i] = vzero;

  for (int t0 = 0; t0 < Tc; t0 += 32) {
    #pragma unroll
    for (int j = 0; j < 4; ++j) {
      const int s = tid + j * 128;              // 512 quad-slots
      const int d = s & 63, tq = (s >> 6) << 2; // tq in {0,4,...,28}
      const float* kp = kbuf + base + (size_t)(t0 + tq) * Dc + d;
      *(v4h*)&ksT[d][tq] = pk4(kp[0], kp[Dc], kp[2 * Dc], kp[3 * Dc]);
      const float* vp = vbuf + base + (size_t)(t0 + tq) * Dc + d;
      *(v4h*)&vsT[d][tq] = pk4(vp[0], vp[Dc], vp[2 * Dc], vp[3 * Dc]);
    }
    __syncthreads();

    const int kb  = khalf * 8;
    const int kb2 = khalf * 16;
    const int m = w * 16 + mlo;
    const v16h af = make_frag(&ksT[m][kb], &ksT[m][kb + 16]);   // A[d][t]
    v16h bf[4];
    #pragma unroll
    for (int nt = 0; nt < 4; ++nt) {
      const int n = nt * 16 + mlo;
      bf[nt] = make_frag(&vsT[n][kb], &vsT[n][kb + 16]);        // B[t][n]
    }
    #pragma unroll
    for (int nt = 0; nt < 4; ++nt)
      acc[nt] = __builtin_amdgcn_wmma_f32_16x16x32_f16(
          false, af, false, bf[nt], (short)0, acc[nt], false, false);
    __syncthreads();
  }
  #pragma unroll
  for (int nt = 0; nt < 4; ++nt)
    #pragma unroll
    for (int r = 0; r < 8; ++r) {
      const int rowd = w * 16 + khalf * 8 + r;
      const int col  = nt * 16 + mlo;
      kvm[(size_t)bh * 4096 + rowd * 64 + col] = acc[nt][r];
    }
}

// ---------------------------------------------------------------------------
// y[b,t,h,:] = (q[b,t,h,:] @ kvm[b,h]) * z[b,h,t]   (y written in f16)
// grid = (Tc/128, B*H), 128 threads (4 waves). kvm cached transposed [n][d];
// q fragments straight from global with 4x b128 loads + pkrtz.
// ---------------------------------------------------------------------------
__global__ __launch_bounds__(128) void attn_apply_kernel(const float* __restrict__ qbuf,
                                                         const float* __restrict__ kvm,
                                                         const float* __restrict__ zbuf,
                                                         _Float16* __restrict__ ybuf)
{
  __shared__ _Float16 kvsT[64][72];   // [n][d]
  const int bh = blockIdx.y;
  const int b  = bh / Hc, h = bh % Hc;
  const int tid = threadIdx.x, lane = tid & 31, w = tid >> 5;
  const int mlo = lane & 15, khalf = lane >> 4;

  #pragma unroll
  for (int j = 0; j < 8; ++j) {
    const int s = tid + j * 128;               // 1024 quad-slots
    const int n = s & 63, dq = (s >> 6) << 2;  // dq in {0,4,...,60}
    const float* p = kvm + (size_t)bh * 4096 + (size_t)dq * 64 + n;
    *(v4h*)&kvsT[n][dq] = pk4(p[0], p[64], p[128], p[192]);
  }
  __syncthreads();

  const int t0 = blockIdx.x * 128 + w * 32;
  const v8f vzero = {0.f,0.f,0.f,0.f,0.f,0.f,0.f,0.f};
  v8f acc[2][4];
  #pragma unroll
  for (int i = 0; i < 2; ++i)
    #pragma unroll
    for (int j = 0; j < 4; ++j) acc[i][j] = vzero;

  const int kb  = khalf * 8;
  const int kb2 = khalf * 16;
  #pragma unroll
  for (int kk = 0; kk < 64; kk += 32) {
    v16h af[2];
    #pragma unroll
    for (int mt = 0; mt < 2; ++mt) {
      const size_t qrow = ((size_t)b * Tc + t0 + mt * 16 + mlo) * Dc + h * HD + kk + kb;
      const float4 q0 = *(const float4*)(qbuf + qrow);
      const float4 q1 = *(const float4*)(qbuf + qrow + 4);
      const float4 q2 = *(const float4*)(qbuf + qrow + 16);
      const float4 q3 = *(const float4*)(qbuf + qrow + 20);
      const v4h h0 = pk4(q0.x, q0.y, q0.z, q0.w);
      const v4h h1 = pk4(q1.x, q1.y, q1.z, q1.w);
      const v4h h2 = pk4(q2.x, q2.y, q2.z, q2.w);
      const v4h h3 = pk4(q3.x, q3.y, q3.z, q3.w);
      const v8h lo = __builtin_shufflevector(h0, h1, 0,1,2,3,4,5,6,7);
      const v8h hi = __builtin_shufflevector(h2, h3, 0,1,2,3,4,5,6,7);
      af[mt] = __builtin_shufflevector(lo, hi, 0,1,2,3,4,5,6,7,8,9,10,11,12,13,14,15);
    }
    v16h bf[4];
    #pragma unroll
    for (int nt = 0; nt < 4; ++nt) {
      const int n = nt * 16 + mlo;
      bf[nt] = make_frag(&kvsT[n][kk + kb], &kvsT[n][kk + kb + 16]);
    }
    #pragma unroll
    for (int nt = 0; nt < 4; ++nt)
      #pragma unroll
      for (int mt = 0; mt < 2; ++mt)
        acc[mt][nt] = __builtin_amdgcn_wmma_f32_16x16x32_f16(
            false, af[mt], false, bf[nt], (short)0, acc[mt][nt], false, false);
  }
  #pragma unroll
  for (int mt = 0; mt < 2; ++mt)
    #pragma unroll
    for (int nt = 0; nt < 4; ++nt)
      #pragma unroll
      for (int r = 0; r < 8; ++r) {
        const int t   = t0 + mt * 16 + khalf * 8 + r;
        const int col = nt * 16 + mlo;
        const float zv = zbuf[(size_t)bh * Tc + t];
        ybuf[((size_t)b * Tc + t) * Dc + h * HD + col] = (_Float16)(acc[mt][nt][r] * zv);
      }
}

// ---------------------------------------------------------------------------
// LayerNorm over D=768, one block per token (256 threads, 3 elems each).
// Reads f32 residual, writes f16 activations (GEMM A-operand).
// ---------------------------------------------------------------------------
__global__ __launch_bounds__(256) void layernorm_kernel(const float* __restrict__ x,
                                                        const float* __restrict__ g,
                                                        const float* __restrict__ be,
                                                        _Float16* __restrict__ out)
{
  __shared__ float red[256];
  const int row = blockIdx.x, tid = threadIdx.x;
  const float* xr = x + (size_t)row * Dc;
  const float v0 = xr[tid], v1 = xr[tid + 256], v2 = xr[tid + 512];
  red[tid] = v0 + v1 + v2;
  __syncthreads();
  for (int s = 128; s > 0; s >>= 1) { if (tid < s) red[tid] += red[tid + s]; __syncthreads(); }
  const float mean = red[0] * (1.0f / Dc);
  __syncthreads();
  const float d0 = v0 - mean, d1 = v1 - mean, d2 = v2 - mean;
  red[tid] = d0 * d0 + d1 * d1 + d2 * d2;
  __syncthreads();
  for (int s = 128; s > 0; s >>= 1) { if (tid < s) red[tid] += red[tid + s]; __syncthreads(); }
  const float inv = rsqrtf(red[0] * (1.0f / Dc) + 1e-5f);
  _Float16* o = out + (size_t)row * Dc;
  o[tid]       = (_Float16)(d0 * inv * g[tid]       + be[tid]);
  o[tid + 256] = (_Float16)(d1 * inv * g[tid + 256] + be[tid + 256]);
  o[tid + 512] = (_Float16)(d2 * inv * g[tid + 512] + be[tid + 512]);
}

// LN + exact-gelu for the action path (grid = B), f32 out (feeds matvec)
__global__ __launch_bounds__(256) void act_ln_gelu_kernel(const float* __restrict__ x,
                                                          const float* __restrict__ g,
                                                          const float* __restrict__ be,
                                                          float* __restrict__ out)
{
  __shared__ float red[256];
  const int row = blockIdx.x, tid = threadIdx.x;
  const float* xr = x + (size_t)row * Dc;
  const float v0 = xr[tid], v1 = xr[tid + 256], v2 = xr[tid + 512];
  red[tid] = v0 + v1 + v2;
  __syncthreads();
  for (int s = 128; s > 0; s >>= 1) { if (tid < s) red[tid] += red[tid + s]; __syncthreads(); }
  const float mean = red[0] * (1.0f / Dc);
  __syncthreads();
  const float d0 = v0 - mean, d1 = v1 - mean, d2 = v2 - mean;
  red[tid] = d0 * d0 + d1 * d1 + d2 * d2;
  __syncthreads();
  for (int s = 128; s > 0; s >>= 1) { if (tid < s) red[tid] += red[tid + s]; __syncthreads(); }
  const float inv = rsqrtf(red[0] * (1.0f / Dc) + 1e-5f);
  float* o = out + (size_t)row * Dc;
  o[tid]       = gelu_exactf(d0 * inv * g[tid]       + be[tid]);
  o[tid + 256] = gelu_exactf(d1 * inv * g[tid + 256] + be[tid + 256]);
  o[tid + 512] = gelu_exactf(d2 * inv * g[tid + 512] + be[tid + 512]);
}

// Small matvec for 8-row action path: out[row] = act(in[row] @ W + bias)
__global__ __launch_bounds__(256) void matvec_kernel(const float* __restrict__ in,
                                                     const float* __restrict__ W,
                                                     const float* __restrict__ bias,
                                                     float* __restrict__ out,
                                                     int Kd, int Nd, int act)
{
  const int row = blockIdx.y;
  const int col = blockIdx.x * 256 + threadIdx.x;
  float s = bias[col];
  const float* ir = in + (size_t)row * Kd;
  for (int k = 0; k < Kd; ++k) s += ir[k] * W[(size_t)k * Nd + col];
  if (act) s = elu1f(s);
  out[(size_t)row * Nd + col] = s;
}

// x = concat(context_latents, mq[target_indices]), one block per token
__global__ __launch_bounds__(256) void gather_kernel(const float* __restrict__ ctx,
                                                     const float* __restrict__ mq,
                                                     const int* __restrict__ idx,
                                                     float* __restrict__ x)
{
  const int tok = blockIdx.x;
  const int b = tok / Tc, t = tok % Tc;
  const float* src = (t < CLc) ? (ctx + ((size_t)b * CLc + t) * Dc)
                               : (mq + (size_t)idx[b * TTc + (t - CLc)] * Dc);
  float* dst = x + (size_t)tok * Dc;
  dst[threadIdx.x]       = src[threadIdx.x];
  dst[threadIdx.x + 256] = src[threadIdx.x + 256];
  dst[threadIdx.x + 512] = src[threadIdx.x + 512];
}

// Cross-attn (kv length 1): kvm = outer(k,v); ksum = k. grid = B*H, 64 threads
__global__ __launch_bounds__(64) void cross_outer_kernel(const float* __restrict__ kc,
                                                         const float* __restrict__ vc,
                                                         float* __restrict__ kvm,
                                                         float* __restrict__ ksum)
{
  const int bh = blockIdx.x;
  const int b = bh / Hc, h = bh % Hc;
  const int e = threadIdx.x;
  const float ve = vc[(size_t)b * Dc + h * HD + e];
  ksum[(size_t)bh * HD + e] = kc[(size_t)b * Dc + h * HD + e];
  for (int d = 0; d < HD; ++d)
    kvm[(size_t)bh * 4096 + d * 64 + e] = kc[(size_t)b * Dc + h * HD + d] * ve;
}

// ksum[b,h,d] = sum_t k[b,t,h,d].  grid = B*H, 256 threads (4 chunks x 64 d)
__global__ __launch_bounds__(256) void ksum_kernel(const float* __restrict__ kbuf,
                                                   float* __restrict__ ksum)
{
  __shared__ float red[256];
  const int bh = blockIdx.x;
  const int b = bh / Hc, h = bh % Hc;
  const int tid = threadIdx.x;
  const int d = tid & 63, c = tid >> 6;
  float s = 0.f;
  for (int t = c; t < Tc; t += 4)
    s += kbuf[((size_t)b * Tc + t) * Dc + h * HD + d];
  red[tid] = s;
  __syncthreads();
  if (c == 0) ksum[(size_t)bh * HD + d] = red[d] + red[64 + d] + red[128 + d] + red[192 + d];
}

// z[b,h,t] = 1 / (q[b,t,h,:] . ksum[b,h,:] + 1e-6)
__global__ __launch_bounds__(256) void z_kernel(const float* __restrict__ qbuf,
                                                const float* __restrict__ ksum,
                                                float* __restrict__ zbuf)
{
  const size_t i = (size_t)blockIdx.x * 256 + threadIdx.x;  // over B*H*Tc
  const int bh = (int)(i / Tc), t = (int)(i % Tc);
  const int b = bh / Hc, h = bh % Hc;
  float s = 1e-6f;
  const float* q = qbuf + ((size_t)b * Tc + t) * Dc + h * HD;
  const float* ks = ksum + (size_t)bh * HD;
  #pragma unroll 8
  for (int d = 0; d < HD; ++d) s += q[d] * ks[d];
  zbuf[i] = 1.0f / s;
}

// pred = xn[:, CL:, :] contiguously (f16), one block per pred row
__global__ __launch_bounds__(256) void pred_copy_kernel(const _Float16* __restrict__ xn,
                                                        _Float16* __restrict__ pred)
{
  const int row = blockIdx.x;
  const int b = row / TTc, t = row % TTc;
  const _Float16* src = xn + ((size_t)b * Tc + CLc + t) * Dc;
  _Float16* dst = pred + (size_t)row * Dc;
  dst[threadIdx.x]       = src[threadIdx.x];
  dst[threadIdx.x + 256] = src[threadIdx.x + 256];
  dst[threadIdx.x + 512] = src[threadIdx.x + 512];
}

// ---------------------------------------------------------------------------
static void launch_gemm_f32(int epi, const _Float16* A, const float* W, const float* bias,
                            float* C, int M, int N, int K, hipStream_t s)
{
  dim3 g(N / 128, M / 128), b(256);
  switch (epi) {
    case 0: gemm_kernel<0, float><<<g, b, 0, s>>>(A, W, bias, C, M, N, K); break;
    case 1: gemm_kernel<1, float><<<g, b, 0, s>>>(A, W, bias, C, M, N, K); break;
    case 3: gemm_kernel<3, float><<<g, b, 0, s>>>(A, W, bias, C, M, N, K); break;
    case 4: gemm_kernel<4, float><<<g, b, 0, s>>>(A, W, bias, C, M, N, K); break;
  }
}
static void launch_gemm_f16_gelu(const _Float16* A, const float* W, const float* bias,
                                 _Float16* C, int M, int N, int K, hipStream_t s)
{
  dim3 g(N / 128, M / 128), b(256);
  gemm_kernel<2, _Float16><<<g, b, 0, s>>>(A, W, bias, C, M, N, K);
}

extern "C" void kernel_launch(void* const* d_in, const int* in_sizes, int n_in,
                              void* d_out, int out_size, void* d_ws, size_t ws_size,
                              hipStream_t stream)
{
  (void)in_sizes; (void)n_in; (void)out_size; (void)ws_size;

  const float* ctx    = (const float*)d_in[0];
  const float* actl   = (const float*)d_in[1];
  const int*   tidx   = (const int*)  d_in[2];
  const float* ad_w1  = (const float*)d_in[3];
  const float* ad_b1  = (const float*)d_in[4];
  const float* ad_lnw = (const float*)d_in[5];
  const float* ad_lnb = (const float*)d_in[6];
  const float* ad_w2  = (const float*)d_in[7];
  const float* ad_b2  = (const float*)d_in[8];
  const float* mq     = (const float*)d_in[9];
  const float* ln1_w  = (const float*)d_in[10];
  const float* ln1_b  = (const float*)d_in[11];
  const float* a_qw   = (const float*)d_in[12];
  const float* a_qb   = (const float*)d_in[13];
  const float* a_kw   = (const float*)d_in[14];
  const float* a_kb   = (const float*)d_in[15];
  const float* a_vw   = (const float*)d_in[16];
  const float* a_vb   = (const float*)d_in[17];
  const float* a_cw   = (const float*)d_in[18];
  const float* a_cb   = (const float*)d_in[19];
  const float* ln2_w  = (const float*)d_in[20];
  const float* ln2_b  = (const float*)d_in[21];
  const float* s_qw   = (const float*)d_in[22];
  const float* s_qb   = (const float*)d_in[23];
  const float* s_kw   = (const float*)d_in[24];
  const float* s_kb   = (const float*)d_in[25];
  const float* s_vw   = (const float*)d_in[26];
  const float* s_vb   = (const float*)d_in[27];
  const float* s_cw   = (const float*)d_in[28];
  const float* s_cb   = (const float*)d_in[29];
  const float* ln3_w  = (const float*)d_in[30];
  const float* ln3_b  = (const float*)d_in[31];
  const float* mlp_w1 = (const float*)d_in[32];
  const float* mlp_b1 = (const float*)d_in[33];
  const float* mlp_w2 = (const float*)d_in[34];
  const float* mlp_b2 = (const float*)d_in[35];
  const float* fn_w   = (const float*)d_in[36];
  const float* fn_b   = (const float*)d_in[37];
  const float* mu_w   = (const float*)d_in[38];
  const float* mu_b   = (const float*)d_in[39];
  const float* lv_w   = (const float*)d_in[40];
  const float* lv_b   = (const float*)d_in[41];

  float* ws = (float*)d_ws;
  size_t off = 0;
  auto salloc = [&](size_t nfloats) {
    float* p = ws + off; off += (nfloats + 63) & ~(size_t)63; return p;
  };

  float*    x      = salloc((size_t)NTOK * Dc);
  _Float16* xn     = (_Float16*)salloc((size_t)NTOK * Dc / 2);       // f16
  float*    qb     = salloc((size_t)NTOK * Dc);
  float*    kb     = salloc((size_t)NTOK * Dc);
  float*    vb     = salloc((size_t)NTOK * Dc);
  _Float16* yb     = (_Float16*)salloc((size_t)NTOK * Dc / 2);       // f16
  _Float16* hb     = (_Float16*)salloc((size_t)NTOK * Fc / 2);       // f16
  _Float16* pred   = (_Float16*)salloc((size_t)Bc * TTc * Dc / 2);   // f16
  float*    kvm    = salloc((size_t)Bc * Hc * HD * HD);
  float*    ksum   = salloc((size_t)Bc * Hc * HD);
  float*    zb     = salloc((size_t)Bc * Hc * Tc);
  float*    a1     = salloc((size_t)Bc * Dc);
  float*    a2     = salloc((size_t)Bc * Dc);
  float*    ae     = salloc((size_t)Bc * Dc);
  float*    kc     = salloc((size_t)Bc * Dc);
  float*    vc     = salloc((size_t)Bc * Dc);

  float* out_mu = (float*)d_out;
  float* out_lv = out_mu + (size_t)Bc * TTc * Dc;

  const dim3 tok_grid(NTOK), blk256(256);
  const dim3 mv_grid(Dc / 256, Bc);
  const dim3 bh_grid(Bc * Hc);
  const dim3 apply_grid(Tc / 128, Bc * Hc);
  const int  zblocks = (Bc * Hc * Tc) / 256;

  // x = concat(context, mq[idx])
  gather_kernel<<<tok_grid, blk256, 0, stream>>>(ctx, mq, tidx, x);

  // action embedding (8 rows — tiny, VALU path)
  matvec_kernel<<<mv_grid, blk256, 0, stream>>>(actl, ad_w1, ad_b1, a1, Acn, Dc, 0);
  act_ln_gelu_kernel<<<dim3(Bc), blk256, 0, stream>>>(a1, ad_lnw, ad_lnb, a2);
  matvec_kernel<<<mv_grid, blk256, 0, stream>>>(a2, ad_w2, ad_b2, ae, Dc, Dc, 0);

  for (int i = 0; i < Lc; ++i) {
    const size_t wD = (size_t)i * Dc * Dc, bD = (size_t)i * Dc;
    const size_t wF = (size_t)i * Dc * Fc, bF = (size_t)i * Fc;

    // ---- cross linear attention (kv length 1) ----
    layernorm_kernel<<<tok_grid, blk256, 0, stream>>>(x, ln1_w + bD, ln1_b + bD, xn);
    matvec_kernel<<<mv_grid, blk256, 0, stream>>>(ae, a_kw + wD, a_kb + bD, kc, Dc, Dc, 1);
    matvec_kernel<<<mv_grid, blk256, 0, stream>>>(ae, a_vw + wD, a_vb + bD, vc, Dc, Dc, 0);
    cross_outer_kernel<<<bh_grid, dim3(64), 0, stream>>>(kc, vc, kvm, ksum);
    launch_gemm_f32(1, xn, a_qw + wD, a_qb + bD, qb, NTOK, Dc, Dc, stream);  // q=elu+1
    z_kernel<<<dim3(zblocks), blk256, 0, stream>>>(qb, ksum, zb);
    attn_apply_kernel<<<apply_grid, dim3(128), 0, stream>>>(qb, kvm, zb, yb);
    launch_gemm_f32(3, yb, a_cw + wD, a_cb + bD, x, NTOK, Dc, Dc, stream);   // x += c-proj

    // ---- self linear attention ----
    layernorm_kernel<<<tok_grid, blk256, 0, stream>>>(x, ln2_w + bD, ln2_b + bD, xn);
    launch_gemm_f32(1, xn, s_qw + wD, s_qb + bD, qb, NTOK, Dc, Dc, stream);  // q=elu+1
    launch_gemm_f32(1, xn, s_kw + wD, s_kb + bD, kb, NTOK, Dc, Dc, stream);  // k=elu+1
    launch_gemm_f32(0, xn, s_vw + wD, s_vb + bD, vb, NTOK, Dc, Dc, stream);  // v
    attn_kv_kernel<<<bh_grid, dim3(128), 0, stream>>>(kb, vb, kvm);
    ksum_kernel<<<bh_grid, blk256, 0, stream>>>(kb, ksum);
    z_kernel<<<dim3(zblocks), blk256, 0, stream>>>(qb, ksum, zb);
    attn_apply_kernel<<<apply_grid, dim3(128), 0, stream>>>(qb, kvm, zb, yb);
    launch_gemm_f32(3, yb, s_cw + wD, s_cb + bD, x, NTOK, Dc, Dc, stream);   // x += c-proj

    // ---- MLP ----
    layernorm_kernel<<<tok_grid, blk256, 0, stream>>>(x, ln3_w + bD, ln3_b + bD, xn);
    launch_gemm_f16_gelu(xn, mlp_w1 + wF, mlp_b1 + bF, hb, NTOK, Fc, Dc, stream);
    launch_gemm_f32(3, hb, mlp_w2 + wF, mlp_b2 + bD, x, NTOK, Dc, Fc, stream); // x += mlp
  }

  // final norm + heads
  layernorm_kernel<<<tok_grid, blk256, 0, stream>>>(x, fn_w, fn_b, xn);
  pred_copy_kernel<<<dim3(Bc * TTc), blk256, 0, stream>>>(xn, pred);
  launch_gemm_f32(0, pred, mu_w, mu_b, out_mu, Bc * TTc, Dc, Dc, stream);
  launch_gemm_f32(4, pred, lv_w, lv_b, out_lv, Bc * TTc, Dc, Dc, stream);    // clip(-10,2)
}
